// MambaLayer_53876069761556
// MI455X (gfx1250) — matrix operand order repaired and checked by reading, not compiled, not run on hardware
//
#include <hip/hip_runtime.h>
#include <hip/hip_bf16.h>

// ---------------------------------------------------------------------------
// Mamba layer for MI455X (gfx1250, wave32, WMMA).
// GEMMs use v_wmma_f32_16x16x32_f16 (f16 in / f32 accumulate) with explicit
// double-buffered fragments so next-step VMEM overlaps current-step WMMAs.
// Requires N % 64 == 0 for all GEMMs (true here: 2048/64/1024/512).
// ---------------------------------------------------------------------------

typedef __attribute__((ext_vector_type(16))) _Float16 v16h;
typedef __attribute__((ext_vector_type(8)))  _Float16 v8h;
typedef __attribute__((ext_vector_type(8)))  float    v8f;

#define MROWS   8192      // B*L
#define DMODEL  512
#define DINNER  1024
#define DSTATE  16
#define DTRANK  32
#define DFF     2048
#define SEQL    1024

// ---------------------------------------------------------------- cast f32->f16
__global__ __launch_bounds__(256) void cast_f16_kernel(const float* __restrict__ in,
                                                       _Float16* __restrict__ out, int n) {
    int i = blockIdx.x * 256 + threadIdx.x;
    if (i < n) out[i] = (_Float16)in[i];
}

// ---------------------------------------------------------------- WMMA GEMM
// C[M,N] = A[M,K](f16, row stride lda) * B[N,K]^T (f16, row stride ldb)
// EPI: 0=none, 1=+bias, 2=+bias then exact GELU, 3=+bias then softplus
// Wave tile: 32(M) x 64(N) -> 2 A frags x 4 B frags = 8 WMMAs per k-step.
// Block: 8 waves stacked in M (256 rows). Branch-free WMMA region.
union V16U { v16h v; v8h h[2]; };

__device__ __forceinline__ void load_frag_a(const _Float16* __restrict__ row, int k0,
                                            int hlf, V16U& f) {
    // A fragment (16x32): lanes 0-15 -> K {k0..+7, k0+16..+23},
    //                     lanes 16-31 -> K {k0+8..+15, k0+24..+31}
    f.h[0] = *(const v8h*)(row + k0 + 8 * hlf);
    f.h[1] = *(const v8h*)(row + k0 + 16 + 8 * hlf);
}
__device__ __forceinline__ void load_frag_b(const _Float16* __restrict__ row, int k0,
                                            int hlf, V16U& f) {
    // B fragment (32x16): lanes 0-15 -> K {k0..+7, k0+16..+23}, lanes 16-31 -> K {k0+8..+15, k0+24..+31}
    f.h[0] = *(const v8h*)(row + k0 + 8 * hlf);
    f.h[1] = *(const v8h*)(row + k0 + 16 + 8 * hlf);
}

template<int EPI>
__global__ __launch_bounds__(256) void wmma_gemm(const _Float16* __restrict__ A, int lda,
                                                 const _Float16* __restrict__ Bw, int ldb,
                                                 float* __restrict__ C,
                                                 _Float16* __restrict__ C16,
                                                 const float* __restrict__ bias,
                                                 int N, int K) {
    const int wave  = threadIdx.x >> 5;
    const int lane  = threadIdx.x & 31;
    const int hlf   = lane >> 4;       // 0 -> lanes 0-15, 1 -> lanes 16-31
    const int lrow  = lane & 15;
    const int m0    = (blockIdx.x * 8 + wave) * 32;
    const int n0    = blockIdx.y * 64;

    v8f acc[2][4] = {};

    const _Float16* arow0 = A + (size_t)(m0 + lrow) * lda;
    const _Float16* arow1 = arow0 + (size_t)16 * lda;
    const _Float16* brow[4];
    #pragma unroll
    for (int t = 0; t < 4; ++t)
        brow[t] = Bw + (size_t)(n0 + t * 16 + lrow) * ldb;

    auto LOAD = [&](int k0, V16U& A0, V16U& A1, V16U* Bf) {
        load_frag_a(arow0, k0, hlf, A0);
        load_frag_a(arow1, k0, hlf, A1);
        #pragma unroll
        for (int t = 0; t < 4; ++t) load_frag_b(brow[t], k0, hlf, Bf[t]);
        // speculative prefetch two steps ahead (silently dropped if OOB)
        __builtin_prefetch(arow0 + k0 + 64, 0, 3);
        __builtin_prefetch(arow1 + k0 + 64, 0, 3);
    };
    auto MMA = [&](const V16U& A0, const V16U& A1, const V16U* Bf) {
        #pragma unroll
        for (int t = 0; t < 4; ++t) {
            acc[0][t] = __builtin_amdgcn_wmma_f32_16x16x32_f16(
                            false, A0.v, false, Bf[t].v, (short)0, acc[0][t], false, false);
            acc[1][t] = __builtin_amdgcn_wmma_f32_16x16x32_f16(
                            false, A1.v, false, Bf[t].v, (short)0, acc[1][t], false, false);
        }
    };

    // Ping-pong double buffering: loads of step s+1 issue before WMMAs of step s.
    V16U xa0, xa1, xb[4];
    V16U ya0, ya1, yb[4];
    const int nsteps = K >> 5;           // K/32
    LOAD(0, xa0, xa1, xb);
    for (int s = 0; s + 2 <= nsteps; s += 2) {
        LOAD((s + 1) * 32, ya0, ya1, yb);
        MMA(xa0, xa1, xb);
        if (s + 2 < nsteps) LOAD((s + 2) * 32, xa0, xa1, xb);
        MMA(ya0, ya1, yb);
    }
    if (nsteps & 1) MMA(xa0, xa1, xb);   // odd tail (dt_proj: K=32 -> 1 step)

    // Epilogue. f32 C/D layout: VGPR r -> row (r + 8*hlf), col = lrow within tile.
    #pragma unroll
    for (int i = 0; i < 2; ++i) {
        #pragma unroll
        for (int t = 0; t < 4; ++t) {
            int col = n0 + t * 16 + lrow;
            float bv = (EPI >= 1) ? bias[col] : 0.0f;
            #pragma unroll
            for (int r = 0; r < 8; ++r) {
                int m = m0 + i * 16 + r + 8 * hlf;
                float v = acc[i][t][r];
                if (EPI >= 1) v += bv;
                if (EPI == 2) v = 0.5f * v * (1.0f + erff(v * 0.70710678118654752f)); // GELU
                if (EPI == 3) v = fmaxf(v, 0.0f) + log1pf(expf(-fabsf(v)));          // softplus
                size_t off = (size_t)m * N + col;
                if (C)   C[off]   = v;
                if (C16) C16[off] = (_Float16)v;
            }
        }
    }
}

// ---------------------------------------------------------------- depthwise conv + SiLU
__global__ __launch_bounds__(256) void conv_silu_kernel(const float* __restrict__ xz,
                                                        const float* __restrict__ cw,
                                                        const float* __restrict__ cb,
                                                        float* __restrict__ u,
                                                        _Float16* __restrict__ u16) {
    int idx = blockIdx.x * 256 + threadIdx.x;        // over MROWS*DINNER
    int d   = idx & (DINNER - 1);
    int row = idx >> 10;
    int l   = row & (SEQL - 1);
    int b   = row >> 10;
    float acc = cb[d];
    #pragma unroll
    for (int j = 0; j < 4; ++j) {
        int ll = l - 3 + j;
        if (ll >= 0)
            acc += cw[d * 4 + j] * xz[((size_t)(b * SEQL + ll)) * (2 * DINNER) + d];
    }
    float s = acc / (1.0f + expf(-acc));             // SiLU
    u[idx]   = s;
    u16[idx] = (_Float16)s;
}

// ---------------------------------------------------------------- selective scan
// one thread per (b, d) channel; 16-state recurrence kept in VGPRs.
__global__ __launch_bounds__(256) void scan_kernel(const float* __restrict__ dt,
                                                   const float* __restrict__ u,
                                                   const float* __restrict__ xdbl,
                                                   const float* __restrict__ xz,
                                                   const float* __restrict__ A_log,
                                                   const float* __restrict__ Dv,
                                                   _Float16* __restrict__ y16) {
    int t = blockIdx.x * 256 + threadIdx.x;          // 0 .. 8191
    int d = t & (DINNER - 1);
    int b = t >> 10;
    float a[DSTATE], h[DSTATE];
    #pragma unroll
    for (int s = 0; s < DSTATE; ++s) { a[s] = -expf(A_log[d * DSTATE + s]); h[s] = 0.0f; }
    float Dd = Dv[d];
    for (int l = 0; l < SEQL; ++l) {
        size_t row = (size_t)b * SEQL + l;
        float dtv = dt[row * DINNER + d];
        float uv  = u[row * DINNER + d];
        const float* Bc = xdbl + row * 64 + DTRANK;          // cols 32..47
        const float* Cc = xdbl + row * 64 + DTRANK + DSTATE; // cols 48..63
        float yv = 0.0f;
        #pragma unroll
        for (int s = 0; s < DSTATE; ++s) {
            float dA = expf(dtv * a[s]);
            h[s] = h[s] * dA + dtv * Bc[s] * uv;
            yv  += h[s] * Cc[s];
        }
        float zv = xz[row * (2 * DINNER) + DINNER + d];
        float g  = zv / (1.0f + expf(-zv));          // SiLU gate
        y16[row * DINNER + d] = (_Float16)((yv + uv * Dd) * g);
    }
}

// ---------------------------------------------------------------- residual + LayerNorm (512 cols)
__global__ __launch_bounds__(256) void add_ln512_kernel(const float* __restrict__ xa,
                                                        const float* __restrict__ xb,
                                                        const float* __restrict__ g,
                                                        const float* __restrict__ bb,
                                                        float* __restrict__ out,
                                                        _Float16* __restrict__ out16) {
    __shared__ float ssum[256];
    __shared__ float ssq[256];
    int row = blockIdx.x;
    int t   = threadIdx.x;
    size_t base = (size_t)row * DMODEL;
    float v0 = xa[base + t]       + xb[base + t];
    float v1 = xa[base + t + 256] + xb[base + t + 256];
    ssum[t] = v0 + v1;
    ssq[t]  = v0 * v0 + v1 * v1;
    __syncthreads();
    for (int off = 128; off > 0; off >>= 1) {
        if (t < off) { ssum[t] += ssum[t + off]; ssq[t] += ssq[t + off]; }
        __syncthreads();
    }
    float mean = ssum[0] * (1.0f / DMODEL);
    float var  = ssq[0]  * (1.0f / DMODEL) - mean * mean;   // population variance (jnp.var)
    float rs   = rsqrtf(var + 1e-12f);
    float o0 = (v0 - mean) * rs * g[t]       + bb[t];
    float o1 = (v1 - mean) * rs * g[t + 256] + bb[t + 256];
    out[base + t]       = o0;
    out[base + t + 256] = o1;
    if (out16) { out16[base + t] = (_Float16)o0; out16[base + t + 256] = (_Float16)o1; }
}

// ---------------------------------------------------------------- launcher
static inline char* carve(char*& p, size_t bytes) {
    char* r = p;
    p += (bytes + 255) & ~(size_t)255;
    return r;
}

extern "C" void kernel_launch(void* const* d_in, const int* in_sizes, int n_in,
                              void* d_out, int out_size, void* d_ws, size_t ws_size,
                              hipStream_t stream) {
    const float* x        = (const float*)d_in[0];
    const float* in_w     = (const float*)d_in[1];
    const float* conv_w   = (const float*)d_in[2];
    const float* conv_b   = (const float*)d_in[3];
    const float* xproj_w  = (const float*)d_in[4];
    const float* dtproj_w = (const float*)d_in[5];
    const float* dtproj_b = (const float*)d_in[6];
    const float* A_log    = (const float*)d_in[7];
    const float* Dv       = (const float*)d_in[8];
    const float* out_w    = (const float*)d_in[9];
    const float* ln1_g    = (const float*)d_in[10];
    const float* ln1_b    = (const float*)d_in[11];
    const float* fc1_w    = (const float*)d_in[12];
    const float* fc1_b    = (const float*)d_in[13];
    const float* fc2_w    = (const float*)d_in[14];
    const float* fc2_b    = (const float*)d_in[15];
    const float* ln2_g    = (const float*)d_in[16];
    const float* ln2_b    = (const float*)d_in[17];
    float* outp = (float*)d_out;

    const size_t M = MROWS;
    char* p = (char*)d_ws;
    // f32 intermediates
    float* xz   = (float*)carve(p, M * 2 * DINNER * 4);
    float* u    = (float*)carve(p, M * DINNER * 4);
    float* xdbl = (float*)carve(p, M * 64 * 4);
    float* dtb  = (float*)carve(p, M * DINNER * 4);
    float* mo   = (float*)carve(p, M * DMODEL * 4);
    float* h1   = (float*)carve(p, M * DMODEL * 4);
    float* f2   = (float*)carve(p, M * DMODEL * 4);
    // f16 intermediates
    _Float16* x16    = (_Float16*)carve(p, M * DMODEL * 2);
    _Float16* u16    = (_Float16*)carve(p, M * DINNER * 2);
    _Float16* xdbl16 = (_Float16*)carve(p, M * 64 * 2);
    _Float16* y16    = (_Float16*)carve(p, M * DINNER * 2);
    _Float16* h16    = (_Float16*)carve(p, M * DMODEL * 2);
    _Float16* fb16   = (_Float16*)carve(p, M * DFF * 2);
    // f16 weights
    _Float16* in_w16  = (_Float16*)carve(p, (size_t)2 * DINNER * DMODEL * 2);
    _Float16* xp_w16  = (_Float16*)carve(p, (size_t)64 * DINNER * 2);
    _Float16* dt_w16  = (_Float16*)carve(p, (size_t)DINNER * DTRANK * 2);
    _Float16* out_w16 = (_Float16*)carve(p, (size_t)DMODEL * DINNER * 2);
    _Float16* fc1_w16 = (_Float16*)carve(p, (size_t)DFF * DMODEL * 2);
    _Float16* fc2_w16 = (_Float16*)carve(p, (size_t)DMODEL * DFF * 2);

    #define CAST(src, dst, n) cast_f16_kernel<<<((n) + 255) / 256, 256, 0, stream>>>(src, dst, (int)(n))
    CAST(x,        x16,     M * DMODEL);
    CAST(in_w,     in_w16,  (size_t)2 * DINNER * DMODEL);
    CAST(xproj_w,  xp_w16,  (size_t)64 * DINNER);
    CAST(dtproj_w, dt_w16,  (size_t)DINNER * DTRANK);
    CAST(out_w,    out_w16, (size_t)DMODEL * DINNER);
    CAST(fc1_w,    fc1_w16, (size_t)DFF * DMODEL);
    CAST(fc2_w,    fc2_w16, (size_t)DMODEL * DFF);

    const int GX = (int)(M / 256);     // 32 blocks: 8 waves x 32 rows each
    #define GEMM(EPI, A_, lda_, B_, ldb_, C_, C16_, bias_, N_, K_)                        \
        wmma_gemm<EPI><<<dim3(GX, (N_) / 64), 256, 0, stream>>>(                          \
            A_, lda_, B_, ldb_, C_, C16_, bias_, N_, K_)

    // 1) in_proj: xz[M,2048] = x16[M,512] * in_w16[2048,512]^T
    GEMM(0, x16, DMODEL, in_w16, DMODEL, xz, (_Float16*)nullptr, (const float*)nullptr,
         2 * DINNER, DMODEL);
    // 2) depthwise causal conv + SiLU  -> u (f32 + f16)
    conv_silu_kernel<<<(int)(M * DINNER / 256), 256, 0, stream>>>(xz, conv_w, conv_b, u, u16);
    // 3) x_proj: xdbl[M,64] = u16 * xp_w16^T
    GEMM(0, u16, DINNER, xp_w16, DINNER, xdbl, (_Float16*)nullptr, (const float*)nullptr,
         64, DINNER);
    CAST(xdbl, xdbl16, M * 64);
    // 4) dt_proj + softplus: dtb[M,1024] = softplus(xdbl16[:, :32] * dt_w16^T + dt_b)
    GEMM(3, xdbl16, 64, dt_w16, DTRANK, dtb, (_Float16*)nullptr, dtproj_b, DINNER, DTRANK);
    // 5) selective scan (+ u*D, SiLU(z) gate) -> y16
    scan_kernel<<<(int)(M / 256), 256, 0, stream>>>(dtb, u, xdbl, xz, A_log, Dv, y16);
    // 6) out_proj: mo[M,512] = y16 * out_w16^T
    GEMM(0, y16, DINNER, out_w16, DINNER, mo, (_Float16*)nullptr, (const float*)nullptr,
         DMODEL, DINNER);
    // 7) h1 = LayerNorm(mo + x)  (also f16 copy for fc1)
    add_ln512_kernel<<<(int)M, 256, 0, stream>>>(mo, x, ln1_g, ln1_b, h1, h16);
    // 8) fc1 + bias + exact GELU -> fb16 (f16 only; feeds fc2 WMMA directly)
    GEMM(2, h16, DMODEL, fc1_w16, DMODEL, (float*)nullptr, fb16, fc1_b, DFF, DMODEL);
    // 9) fc2 + bias -> f2
    GEMM(1, fb16, DFF, fc2_w16, DFF, f2, (_Float16*)nullptr, fc2_b, DMODEL, DFF);
    // 10) out = LayerNorm(f2 + h1)
    add_ln512_kernel<<<(int)M, 256, 0, stream>>>(f2, h1, ln2_g, ln2_b, outp, (_Float16*)nullptr);

    #undef GEMM
    #undef CAST
    (void)in_sizes; (void)n_in; (void)out_size; (void)ws_size;
}
